// MoLExLinear_18966575579734
// MI455X (gfx1250) — matrix operand order repaired and checked by reading, not compiled, not run on hardware
//
#include <hip/hip_runtime.h>
#include <hip/hip_bf16.h>

// ---------------------------------------------------------------------------
// MoE-LoRA linear for MI455X (gfx1250), wave32, WMMA bf16x3 (fp32-accurate).
//   out = x @ W^T + b + sum_{top2 e} gate_e * (x @ A_e^T) @ B_e^T
// Shapes: N=8192 tokens, D=2048, O=2048, E=8, R=16.
// gfx1250 paths: v_wmma_f32_16x16x32_bf16 main loop, ds_load_b128 fragments,
// global_load_async_to_lds_b128 (ASYNCcnt) staging of the LoRA-B epilogue tile.
// ---------------------------------------------------------------------------

typedef __attribute__((ext_vector_type(16))) __bf16 v16bf;
typedef __attribute__((ext_vector_type(8)))  __bf16 v8bf;
typedef __attribute__((ext_vector_type(8)))  float  v8f;

// Async global->LDS 16B copy, tracked by ASYNCcnt (gfx1250).
// Inline asm: VDST = per-lane LDS byte offset (low 32b of generic LDS pointer),
// VADDR = per-lane 64-bit global address, no SADDR ("off").
static __device__ __forceinline__ void async_load_b128(const void* g, void* l) {
  unsigned loff = (unsigned)(uintptr_t)l;             // low 32b = LDS offset
  unsigned long long ga = (unsigned long long)(uintptr_t)g;
  asm volatile("global_load_async_to_lds_b128 %0, %1, off"
               :: "v"(loff), "v"(ga) : "memory");
}

static __device__ __forceinline__ void wait_asynccnt0() {
#if __has_builtin(__builtin_amdgcn_s_wait_asynccnt)
  __builtin_amdgcn_s_wait_asynccnt(0);
#else
  asm volatile("s_wait_asynccnt 0x0" ::: "memory");
#endif
}

static __device__ __forceinline__ v8f wmma_bf16(v16bf a, v16bf b, v8f c) {
  // 8 args: (neg_a, A, neg_b, B, c_mod, C, reuse_a, reuse_b)
  return __builtin_amdgcn_wmma_f32_16x16x32_bf16(false, a, false, b, (short)0, c,
                                                 false, false);
}

static __device__ __forceinline__ unsigned pack_bf16(float a, float b) {
  unsigned short ha = __builtin_bit_cast(unsigned short, (__bf16)a);
  unsigned short hb = __builtin_bit_cast(unsigned short, (__bf16)b);
  return (unsigned)ha | ((unsigned)hb << 16);
}

// Split 4 f32 into bf16 hi + bf16 residual-lo and store packed (2x b32 each).
static __device__ __forceinline__ void split_store4(__bf16* hiB, __bf16* loB,
                                                    int idx, float4 f) {
  float h0 = (float)(__bf16)f.x, h1 = (float)(__bf16)f.y;
  float h2 = (float)(__bf16)f.z, h3 = (float)(__bf16)f.w;
  *(unsigned*)(hiB + idx)     = pack_bf16(f.x, f.y);
  *(unsigned*)(hiB + idx + 2) = pack_bf16(f.z, f.w);
  *(unsigned*)(loB + idx)     = pack_bf16(f.x - h0, f.y - h1);
  *(unsigned*)(loB + idx + 2) = pack_bf16(f.z - h2, f.w - h3);
}

// Two 16B LDS reads -> one 16-element bf16 fragment.
static __device__ __forceinline__ v16bf load_frag(const __bf16* p, int off2) {
  v8bf a = *(const v8bf*)p;
  v8bf b = *(const v8bf*)(p + off2);
  return __builtin_shufflevector(a, b, 0, 1, 2, 3, 4, 5, 6, 7,
                                 8, 9, 10, 11, 12, 13, 14, 15);
}

#define LDSPAD 40   // bf16 tile row stride 40 (80B = 20 banks, conflict-free)
#define BSTRIDE 20  // lora_B LDS row stride in floats (80B, 16B-aligned rows)

// ---------------------------------------------------------------------------
// Router: logits = x @ gate_w^T + gate_b ; top-2 ; softmax.
// One wave per token, gate_w staged in LDS (dynamic, 64 KB).
// ---------------------------------------------------------------------------
__global__ __launch_bounds__(256)
void molex_route(const float* __restrict__ X, const float* __restrict__ gw,
                 const float* __restrict__ gb, int* __restrict__ ridx,
                 float* __restrict__ rg, int Kdim) {
  extern __shared__ char smem_raw[];
  float* gws = (float*)smem_raw;                       // [8][Kdim]
  const int tid = threadIdx.x;
  const int n4 = (8 * Kdim) / 4;
  for (int i = tid; i < n4; i += 256)
    ((float4*)gws)[i] = ((const float4*)gw)[i];
  __syncthreads();

  const int lane = tid & 31, wid = tid >> 5;
  for (int i = 0; i < 8; ++i) {
    const int tok = blockIdx.x * 64 + wid * 8 + i;
    const float* xp = X + (size_t)tok * Kdim;
    float acc[8] = {0.f, 0.f, 0.f, 0.f, 0.f, 0.f, 0.f, 0.f};
    for (int d = lane; d < Kdim; d += 32) {
      float xv = xp[d];
#pragma unroll
      for (int e = 0; e < 8; ++e) acc[e] += xv * gws[e * Kdim + d];
    }
#pragma unroll
    for (int off = 16; off > 0; off >>= 1)
#pragma unroll
      for (int e = 0; e < 8; ++e) acc[e] += __shfl_xor(acc[e], off, 32);
    if (lane == 0) {
      float lg[8];
#pragma unroll
      for (int e = 0; e < 8; ++e) lg[e] = acc[e] + gb[e];
      int e0 = 0;
#pragma unroll
      for (int e = 1; e < 8; ++e) if (lg[e] > lg[e0]) e0 = e;
      int e1 = (e0 == 0) ? 1 : 0;
#pragma unroll
      for (int e = 0; e < 8; ++e)
        if (e != e0 && lg[e] > lg[e1]) e1 = e;
      float p1  = __expf(lg[e1] - lg[e0]);   // exp(second - max)
      float inv = 1.0f / (1.0f + p1);
      ridx[tok * 2 + 0] = e0;
      ridx[tok * 2 + 1] = e1;
      rg[tok * 2 + 0] = inv;
      rg[tok * 2 + 1] = p1 * inv;
    }
  }
}

// ---------------------------------------------------------------------------
// bf16x3 WMMA GEMM: out[M,Ocols] = X[M,K] @ Wm[Ocols,K]^T (+ epilogue).
// Block tile 128x128, BK=32, 256 threads = 8 waves, wave tile 32x64.
// LORA=false : out = xA workspace (Wm = flattened lora_A, Ocols = 128).
// LORA=true  : lora_B tile DMA'd global->LDS asynchronously at kernel start
//              (overlaps the whole K loop); epilogue adds bias + top-2 gated
//              LoRA delta from LDS.
// ---------------------------------------------------------------------------
template <bool LORA>
__global__ __launch_bounds__(256)
void molex_gemm(const float* __restrict__ X, const float* __restrict__ Wm,
                const float* __restrict__ bias, float* __restrict__ out,
                int Kdim, int Ocols,
                const float* __restrict__ loraB, const float* __restrict__ xA,
                const int* __restrict__ ridx, const float* __restrict__ rg) {
  extern __shared__ char smem_raw[];
  __bf16* As_hi = (__bf16*)smem_raw;            // [128][LDSPAD]
  __bf16* As_lo = As_hi + 128 * LDSPAD;
  __bf16* Ws_hi = As_lo + 128 * LDSPAD;
  __bf16* Ws_lo = Ws_hi + 128 * LDSPAD;
  // dedicated epilogue region (disjoint from main tiles -> async can overlap)
  float* BsL  = (float*)(smem_raw + 4 * 128 * LDSPAD * sizeof(__bf16));
  float* coef = BsL + 8 * 128 * BSTRIDE;        // [128][33]
  int*   eix  = (int*)(coef + 128 * 33);        // [128][2]

  const int tid  = threadIdx.x;
  const int lane = tid & 31;
  const int wid  = tid >> 5;
  const int wm   = wid & 3;       // 4 waves along M (32 rows each)
  const int wn   = wid >> 2;      // 2 waves along N (64 cols each)
  const int m0   = blockIdx.y * 128;
  const int n0   = blockIdx.x * 128;

  if constexpr (LORA) {
    // Kick off async DMA of the full lora_B tile (8 experts x 128 cols x R=16,
    // f32) into LDS; completion checked only at the epilogue (ASYNCcnt).
#pragma unroll
    for (int q = 0; q < 4; ++q) {
      int rowq = tid * 4 + q;                    // 0..1023 = (e, col) rows
      int e = rowq >> 7, cl = rowq & 127;
      const float* src = loraB + ((size_t)e * Ocols + (n0 + cl)) * 16;
      float* dst = BsL + (e * 128 + cl) * BSTRIDE;
#pragma unroll
      for (int r = 0; r < 16; r += 4)
        async_load_b128(src + r, dst + r);
    }
  }

  v8f acc[2][4];
#pragma unroll
  for (int mi = 0; mi < 2; ++mi)
#pragma unroll
    for (int ni = 0; ni < 4; ++ni)
      acc[mi][ni] = v8f{0.f, 0.f, 0.f, 0.f, 0.f, 0.f, 0.f, 0.f};

  // staging assignment: 2 threads per row, 16 f32 each
  const int srow  = tid >> 1;
  const int shalf = tid & 1;
  const float* xrow = X  + (size_t)(m0 + srow) * Kdim + shalf * 16;
  const float* wrow = Wm + (size_t)(n0 + srow) * Kdim + shalf * 16;

  const int la  = lane & 15;
  const int k0a = (lane >> 4) * 8;   // A frag: K {k0a..k0a+7, k0a+16..k0a+23}
  const int k0b = (lane >> 4) * 8;   // B frag: K {k0b..k0b+7, k0b+16..k0b+23}

  for (int kk = 0; kk < Kdim; kk += 32) {
#pragma unroll
    for (int v = 0; v < 4; ++v) {
      float4 fx = *(const float4*)(xrow + kk + v * 4);
      float4 fw = *(const float4*)(wrow + kk + v * 4);
      int c = srow * LDSPAD + shalf * 16 + v * 4;
      split_store4(As_hi, As_lo, c, fx);
      split_store4(Ws_hi, Ws_lo, c, fw);
    }
    __builtin_prefetch(xrow + kk + 32, 0, 1);   // global_prefetch next chunk
    __syncthreads();

    v16bf ah[2], al[2], bh[4], bl[4];
#pragma unroll
    for (int mi = 0; mi < 2; ++mi) {
      int r = (wm * 32 + mi * 16 + la) * LDSPAD + k0a;
      ah[mi] = load_frag(As_hi + r, 16);
      al[mi] = load_frag(As_lo + r, 16);
    }
#pragma unroll
    for (int ni = 0; ni < 4; ++ni) {
      int r = (wn * 64 + ni * 16 + la) * LDSPAD + k0b;
      bh[ni] = load_frag(Ws_hi + r, 16);
      bl[ni] = load_frag(Ws_lo + r, 16);
    }
#pragma unroll
    for (int mi = 0; mi < 2; ++mi)
#pragma unroll
      for (int ni = 0; ni < 4; ++ni) {
        acc[mi][ni] = wmma_bf16(ah[mi], bh[ni], acc[mi][ni]);  // hi*hi
        acc[mi][ni] = wmma_bf16(ah[mi], bl[ni], acc[mi][ni]);  // hi*lo
        acc[mi][ni] = wmma_bf16(al[mi], bh[ni], acc[mi][ni]);  // lo*hi
      }
    __syncthreads();
  }

  if constexpr (LORA) {
    // ---- epilogue: gate-weighted xA coeffs + async-staged lora_B tile ----
    {
      int tl = tid >> 1, hf = tid & 1;           // token-local, expert slot
      int tok = m0 + tl;
      int e   = ridx[tok * 2 + hf];
      float g = rg[tok * 2 + hf];
      eix[tl * 2 + hf] = e;
      const float* xa = xA + (size_t)tok * 128 + e * 16;
      float* cd = coef + tl * 33 + hf * 16;
#pragma unroll
      for (int r = 0; r < 16; ++r) cd[r] = g * xa[r];
    }
    wait_asynccnt0();        // this wave's lora_B DMA complete
    __syncthreads();         // all waves' DMA + coeff stores visible

#pragma unroll
    for (int mi = 0; mi < 2; ++mi)
#pragma unroll
      for (int ni = 0; ni < 4; ++ni) {
        int cl  = wn * 64 + ni * 16 + la;
        int col = n0 + cl;
        float bv = bias[col];
#pragma unroll
        for (int j = 0; j < 8; ++j) {
          int mrow = wm * 32 + mi * 16 + j + ((lane >> 4) << 3);
          int e0 = eix[mrow * 2 + 0], e1 = eix[mrow * 2 + 1];
          const float* c0 = coef + mrow * 33;
          const float* b0 = BsL + (e0 * 128 + cl) * BSTRIDE;
          const float* b1 = BsL + (e1 * 128 + cl) * BSTRIDE;
          float d = 0.f;
#pragma unroll
          for (int r = 0; r < 16; ++r)
            d += c0[r] * b0[r] + c0[16 + r] * b1[r];
          // SCALING == 1.0
          out[(size_t)(m0 + mrow) * Ocols + col] = acc[mi][ni][j] + bv + d;
        }
      }
  } else {
#pragma unroll
    for (int mi = 0; mi < 2; ++mi)
#pragma unroll
      for (int ni = 0; ni < 4; ++ni) {
        int col = n0 + wn * 64 + ni * 16 + la;
#pragma unroll
        for (int j = 0; j < 8; ++j) {
          int mrow = wm * 32 + mi * 16 + j + ((lane >> 4) << 3);
          out[(size_t)(m0 + mrow) * Ocols + col] = acc[mi][ni][j];
        }
      }
  }
}

// ---------------------------------------------------------------------------
extern "C" void kernel_launch(void* const* d_in, const int* in_sizes, int n_in,
                              void* d_out, int out_size, void* d_ws, size_t ws_size,
                              hipStream_t stream) {
  const float* x  = (const float*)d_in[0];  // [N, D]
  const float* W  = (const float*)d_in[1];  // [O, D]
  const float* b  = (const float*)d_in[2];  // [O]
  const float* lA = (const float*)d_in[3];  // [E, R, D] -> [E*R, D]
  const float* lB = (const float*)d_in[4];  // [E, O, R]
  const float* gw = (const float*)d_in[5];  // [E, D]
  const float* gb = (const float*)d_in[6];  // [E]
  float* out = (float*)d_out;

  const int E = in_sizes[6];                 // 8
  const int D = in_sizes[5] / E;             // 2048
  const int N = in_sizes[0] / D;             // 8192
  const int O = in_sizes[1] / D;             // 2048
  const int R = in_sizes[3] / (E * D);       // 16
  const int ER = E * R;                      // 128

  // workspace layout
  float* xA  = (float*)d_ws;                 // [N, E*R]   4 MB
  int* ridx  = (int*)(xA + (size_t)N * ER);  // [N, 2]
  float* rg  = (float*)(ridx + (size_t)N * 2);

  const size_t SMEM_MAIN = (size_t)4 * 128 * LDSPAD * sizeof(__bf16);      // 40 KB
  const size_t SMEM_LORA = SMEM_MAIN +
                           (size_t)(8 * 128 * BSTRIDE + 128 * 33) * 4 +    // Bs+coef
                           (size_t)128 * 2 * 4;                            // eix
                                                                           // ~137.5 KB
  // 1) routing: top-2 experts + softmax gates
  molex_route<<<N / 64, 256, (size_t)E * D * sizeof(float), stream>>>(
      x, gw, gb, ridx, rg, D);

  // 2) xA = x @ lora_A^T for all experts (lora_A flat is [E*R, D] row-major)
  molex_gemm<false><<<dim3(ER / 128, N / 128), 256, SMEM_MAIN, stream>>>(
      x, lA, nullptr, xA, D, ER, nullptr, nullptr, nullptr, nullptr);

  // 3) base GEMM + bias + fused top-2 gated LoRA epilogue
  molex_gemm<true><<<dim3(O / 128, N / 128), 256, SMEM_LORA, stream>>>(
      x, W, b, out, D, O, lB, xA, ridx, rg);
}
